// SpatioTemporalBlock_10548439679110
// MI455X (gfx1250) — matrix operand order repaired and checked by reading, not compiled, not run on hardware
//
#include <hip/hip_runtime.h>
#include <hip/hip_bf16.h>
#include <stdint.h>

// ---------------- problem constants (B=8, T=8, S=196, H=768, NH=12, E=4, IE=768)
#define NTOK   12544      // B*T*S
#define HDIM   768
#define QKVN   2304       // 3*H
#define SLEN   196
#define TLEN   8
#define NHEAD  12
#define HD     64
#define NEXP   4
#define IEDIM  768
#define GUN    1536       // 2*IE
#define NSEQ_S 64         // B*T
#define NSEQ_T 1568       // B*S
#define MAXTILES (NTOK/16 + NEXP)   // 788 upper bound on padded 16-row tiles
#define PERMCAP  (NTOK + 16*NEXP)

// meta layout (ints): [0..3]=counts  [4..8]=paddedBase[5]  [9..12]=cursor  [13]=nTiles
#define M_CNT 0
#define M_BASE 4
#define M_CUR 9
#define M_NT 13

typedef __attribute__((ext_vector_type(16))) __bf16 v16bf;
typedef __attribute__((ext_vector_type(8)))  float  v8f;

__device__ __forceinline__ unsigned short f2bf(float f) {
  unsigned int u = __float_as_uint(f);
  unsigned int r = u + 0x7FFFu + ((u >> 16) & 1u);   // round-to-nearest-even
  return (unsigned short)(r >> 16);
}
__device__ __forceinline__ float bf2f(unsigned short h) {
  return __uint_as_float(((unsigned int)h) << 16);
}

union FragU { uint4 u[2]; v16bf v; };

// A fragment (16x32 bf16): lane<16 holds row m=lane, K chunks [k..k+7] and [k+16..k+23];
// lane>=16 same row (lane&15), chunks [k+8..k+15] and [k+24..k+31]. p pre-offset by ko=(lane>>4)*8.
__device__ __forceinline__ v16bf load_fragA(const unsigned short* p) {
  FragU f; f.u[0] = *(const uint4*)(p); f.u[1] = *(const uint4*)(p + 16); return f.v;
}
// B fragment (32x16 bf16, column n per lane, Bt stored [N][K]): lane<16 holds K chunks [k..k+7]
// and [k+16..k+23]; lane>=16 holds [k+8..k+15] and [k+24..k+31]. p pre-offset by ko=(lane>>4)*8.
__device__ __forceinline__ v16bf load_fragB(const unsigned short* p) {
  FragU f; f.u[0] = *(const uint4*)(p); f.u[1] = *(const uint4*)(p + 16); return f.v;
}
__device__ __forceinline__ v16bf zero_frag() {
  FragU f; f.u[0] = make_uint4(0,0,0,0); f.u[1] = make_uint4(0,0,0,0); return f.v;
}

// ---------------- weight convert + transpose: W[R][C] fp32 -> Wt[C][R] bf16
__global__ void convtr_kernel(const float* __restrict__ src, unsigned short* __restrict__ dst,
                              int R, int C) {
  int gid = blockIdx.x * blockDim.x + threadIdx.x;
  if (gid >= R * C) return;
  int r = gid / C, c = gid % C;
  dst[(size_t)c * R + r] = f2bf(src[gid]);
}

// ---------------- layernorm (one block per row) -> bf16
__global__ __launch_bounds__(256) void ln_kernel(const float* __restrict__ x,
                                                 const float* __restrict__ g,
                                                 const float* __restrict__ b,
                                                 unsigned short* __restrict__ out) {
  __shared__ float sh1[256], sh2[256];
  int row = blockIdx.x;
  const float* xr = x + (size_t)row * HDIM;
  float s = 0.f, s2 = 0.f;
  for (int i = threadIdx.x; i < HDIM; i += 256) { float v = xr[i]; s += v; s2 += v * v; }
  sh1[threadIdx.x] = s; sh2[threadIdx.x] = s2; __syncthreads();
  for (int o = 128; o > 0; o >>= 1) {
    if (threadIdx.x < o) { sh1[threadIdx.x] += sh1[threadIdx.x + o]; sh2[threadIdx.x] += sh2[threadIdx.x + o]; }
    __syncthreads();
  }
  float mean = sh1[0] * (1.f / HDIM);
  float var  = sh2[0] * (1.f / HDIM) - mean * mean;
  float inv  = rsqrtf(var + 1e-6f);
  for (int i = threadIdx.x; i < HDIM; i += 256)
    out[(size_t)row * HDIM + i] = f2bf((xr[i] - mean) * inv * g[i] + b[i]);
}

// ---------------- dense WMMA GEMM: C[M][N] = A_bf16[M][K] x Bt_bf16[N][K]^T + bias
// 8 waves, block tile 128x128, wave tile 32x64 (2 A-frags x 4 B-frags -> 8 WMMA / k-step,
// 12 b128 loads per 8 WMMA = 1.5 loads/WMMA).
// flags: 1 = store bf16, 2 = add fp32 residual (fp32 out)
__global__ __launch_bounds__(256) void gemm_bf16_kernel(const unsigned short* __restrict__ A,
                                                        const unsigned short* __restrict__ Bt,
                                                        const float* __restrict__ bias,
                                                        void* Cout, const float* residual,
                                                        int M, int N, int K, int flags) {
  int lane = threadIdx.x & 31, wave = threadIdx.x >> 5;
  int m0 = blockIdx.x * 128 + (wave >> 1) * 32;   // 4 wave-rows
  int n0 = blockIdx.y * 128 + (wave & 1) * 64;    // 2 wave-cols
  int rsel = lane & 15;
  int koA = (lane >> 4) << 3;
  int koB = (lane >> 4) << 3;
  const unsigned short* Ap0 = A + (size_t)(m0 + rsel) * K + koA;
  const unsigned short* Ap1 = Ap0 + (size_t)16 * K;
  const unsigned short* Bp0 = Bt + (size_t)(n0 + rsel) * K + koB;
  const unsigned short* Bp1 = Bp0 + (size_t)16 * K;
  const unsigned short* Bp2 = Bp0 + (size_t)32 * K;
  const unsigned short* Bp3 = Bp0 + (size_t)48 * K;
  v8f acc[2][4] = {};
  for (int k = 0; k < K; k += 32) {
    v16bf a0 = load_fragA(Ap0 + k);
    v16bf a1 = load_fragA(Ap1 + k);
    v16bf b0 = load_fragB(Bp0 + k);
    v16bf b1 = load_fragB(Bp1 + k);
    v16bf b2 = load_fragB(Bp2 + k);
    v16bf b3 = load_fragB(Bp3 + k);
    acc[0][0] = __builtin_amdgcn_wmma_f32_16x16x32_bf16(false, a0, false, b0, (short)0, acc[0][0], false, false);
    acc[0][1] = __builtin_amdgcn_wmma_f32_16x16x32_bf16(false, a0, false, b1, (short)0, acc[0][1], false, false);
    acc[0][2] = __builtin_amdgcn_wmma_f32_16x16x32_bf16(false, a0, false, b2, (short)0, acc[0][2], false, false);
    acc[0][3] = __builtin_amdgcn_wmma_f32_16x16x32_bf16(false, a0, false, b3, (short)0, acc[0][3], false, false);
    acc[1][0] = __builtin_amdgcn_wmma_f32_16x16x32_bf16(false, a1, false, b0, (short)0, acc[1][0], false, false);
    acc[1][1] = __builtin_amdgcn_wmma_f32_16x16x32_bf16(false, a1, false, b1, (short)0, acc[1][1], false, false);
    acc[1][2] = __builtin_amdgcn_wmma_f32_16x16x32_bf16(false, a1, false, b2, (short)0, acc[1][2], false, false);
    acc[1][3] = __builtin_amdgcn_wmma_f32_16x16x32_bf16(false, a1, false, b3, (short)0, acc[1][3], false, false);
  }
  // C/D layout: col = n + (lane&15); rows = m + j + (lane>=16 ? 8 : 0)
  int rowAdd = (lane >> 4) << 3;
  for (int mi = 0; mi < 2; mi++) {
    for (int ni = 0; ni < 4; ni++) {
      int col = n0 + ni * 16 + (lane & 15);
      float bv = bias[col];
      int rbase = m0 + mi * 16 + rowAdd;
      #pragma unroll
      for (int j = 0; j < 8; j++) {
        size_t idx = (size_t)(rbase + j) * N + col;
        float v = acc[mi][ni][j] + bv;
        if (flags & 2) v += residual[idx];
        if (flags & 1) ((unsigned short*)Cout)[idx] = f2bf(v);
        else           ((float*)Cout)[idx] = v;
      }
    }
  }
}

// ---------------- spatial attention: one block per (seq, head), online softmax, LDS K/V (bf16)
__global__ __launch_bounds__(256) void attn_spatial_kernel(const unsigned short* __restrict__ qkv,
                                                           unsigned short* __restrict__ out) {
  __shared__ unsigned short Ks[SLEN * HD];
  __shared__ unsigned short Vs[SLEN * HD];
  int seq = blockIdx.x;   // 0..63  (tokens seq*196 .. +195, contiguous)
  int h   = blockIdx.y;   // 0..11
  size_t base = (size_t)seq * SLEN * QKVN + h * HD;
  for (int e = threadIdx.x; e < SLEN * HD; e += 256) {
    int j = e >> 6, d = e & 63;
    Ks[e] = qkv[base + (size_t)j * QKVN + HDIM     + d];
    Vs[e] = qkv[base + (size_t)j * QKVN + 2 * HDIM + d];
  }
  __syncthreads();
  int q = threadIdx.x;
  if (q < SLEN) {
    float qv[HD];
    #pragma unroll
    for (int d = 0; d < HD; d++) qv[d] = bf2f(qkv[base + (size_t)q * QKVN + d]) * 0.125f;
    float m = -1e30f, l = 0.f, acc[HD];
    #pragma unroll
    for (int d = 0; d < HD; d++) acc[d] = 0.f;
    for (int j = 0; j < SLEN; j++) {
      float s = 0.f;
      #pragma unroll
      for (int d = 0; d < HD; d++) s += qv[d] * bf2f(Ks[j * HD + d]);
      float mn = fmaxf(m, s);
      float sc = __expf(m - mn);
      float p  = __expf(s - mn);
      l = l * sc + p;
      #pragma unroll
      for (int d = 0; d < HD; d++) acc[d] = acc[d] * sc + p * bf2f(Vs[j * HD + d]);
      m = mn;
    }
    float rl = 1.f / l;
    size_t obase = (size_t)(seq * SLEN + q) * HDIM + h * HD;
    #pragma unroll
    for (int d = 0; d < HD; d++) out[obase + d] = f2bf(acc[d] * rl);
  }
}

// ---------------- temporal attention: one block per (b,s); thread -> (head, tq); T=8 keys
__global__ __launch_bounds__(128) void attn_temporal_kernel(const unsigned short* __restrict__ qkv,
                                                            unsigned short* __restrict__ out) {
  int bs = blockIdx.x;                 // 0..1567
  int b = bs / SLEN, s = bs % SLEN;
  int tid = threadIdx.x;
  if (tid >= NHEAD * TLEN) return;
  int h = tid >> 3, tq = tid & 7;
  size_t tokQ = (size_t)b * (TLEN * SLEN) + (size_t)tq * SLEN + s;
  const unsigned short* qp = qkv + tokQ * QKVN + h * HD;
  float qv[HD];
  #pragma unroll
  for (int d = 0; d < HD; d++) qv[d] = bf2f(qp[d]) * 0.125f;
  float m = -1e30f, l = 0.f, acc[HD];
  #pragma unroll
  for (int d = 0; d < HD; d++) acc[d] = 0.f;
  for (int t = 0; t < TLEN; t++) {
    size_t tok = (size_t)b * (TLEN * SLEN) + (size_t)t * SLEN + s;
    const unsigned short* kp = qkv + tok * QKVN + HDIM + h * HD;
    const unsigned short* vp = kp + HDIM;
    float sd = 0.f;
    #pragma unroll
    for (int d = 0; d < HD; d++) sd += qv[d] * bf2f(kp[d]);
    float mn = fmaxf(m, sd);
    float sc = __expf(m - mn);
    float p  = __expf(sd - mn);
    l = l * sc + p;
    #pragma unroll
    for (int d = 0; d < HD; d++) acc[d] = acc[d] * sc + p * bf2f(vp[d]);
    m = mn;
  }
  float rl = 1.f / l;
  size_t obase = tokQ * HDIM + h * HD;
  #pragma unroll
  for (int d = 0; d < HD; d++) out[obase + d] = f2bf(acc[d] * rl);
}

// ---------------- MoE routing helpers
__global__ void moe_init_kernel(int* meta) { if (threadIdx.x < 16) meta[threadIdx.x] = 0; }
__global__ void moe_count_kernel(const int* __restrict__ eids, int* meta) {
  int g = blockIdx.x * blockDim.x + threadIdx.x;
  if (g < NTOK) atomicAdd(&meta[M_CNT + eids[g]], 1);
}
__global__ void moe_scan_kernel(int* meta) {
  if (threadIdx.x == 0 && blockIdx.x == 0) {
    int base = 0;
    meta[M_BASE] = 0;
    for (int e = 0; e < NEXP; e++) {
      base += (meta[M_CNT + e] + 15) & ~15;
      meta[M_BASE + 1 + e] = base;
    }
    meta[M_NT] = base >> 4;
  }
}
__global__ void moe_fill_kernel(int* perm) {
  int g = blockIdx.x * blockDim.x + threadIdx.x;
  if (g < PERMCAP) perm[g] = -1;
}
__global__ void moe_scatter_kernel(const int* __restrict__ eids, int* meta, int* perm) {
  int g = blockIdx.x * blockDim.x + threadIdx.x;
  if (g >= NTOK) return;
  int e = eids[g];
  int pos = atomicAdd(&meta[M_CUR + e], 1);
  perm[meta[M_BASE + e] + pos] = g;
}

// ---------------- grouped (gathered) WMMA GEMM for MoE
// 4 waves (128 thr), tile = 16 gathered rows x 256 cols; wave tile 16x64
// (1 A-frag + 4 B-frags -> 4 WMMA / k-step = 2.5 b128 loads per WMMA). accumulate: +=
__global__ __launch_bounds__(128) void moe_gemm_kernel(const unsigned short* __restrict__ A,
                                                       const unsigned short* __restrict__ BtAll,
                                                       float* Cout,
                                                       const int* __restrict__ perm,
                                                       const int* __restrict__ meta,
                                                       int N, int K, int accumulate) {
  int tile = blockIdx.x;
  if (tile >= meta[M_NT]) return;
  int r16 = tile * 16;
  int e = 0;
  while (e < NEXP - 1 && r16 >= meta[M_BASE + e + 1]) e++;
  int lane = threadIdx.x & 31, wave = threadIdx.x >> 5;
  int n0 = blockIdx.y * 256 + wave * 64;
  int koA = (lane >> 4) << 3;
  int koB = (lane >> 4) << 3;
  int arow = perm[r16 + (lane & 15)];
  const unsigned short* Ap = (arow >= 0) ? (A + (size_t)arow * K + koA) : (const unsigned short*)0;
  const unsigned short* Bp0 = BtAll + ((size_t)e * N + n0 + (lane & 15)) * K + koB;
  const unsigned short* Bp1 = Bp0 + (size_t)16 * K;
  const unsigned short* Bp2 = Bp0 + (size_t)32 * K;
  const unsigned short* Bp3 = Bp0 + (size_t)48 * K;
  v8f acc[4] = {};
  for (int k = 0; k < K; k += 32) {
    v16bf a  = Ap ? load_fragA(Ap + k) : zero_frag();
    v16bf b0 = load_fragB(Bp0 + k);
    v16bf b1 = load_fragB(Bp1 + k);
    v16bf b2 = load_fragB(Bp2 + k);
    v16bf b3 = load_fragB(Bp3 + k);
    acc[0] = __builtin_amdgcn_wmma_f32_16x16x32_bf16(false, a, false, b0, (short)0, acc[0], false, false);
    acc[1] = __builtin_amdgcn_wmma_f32_16x16x32_bf16(false, a, false, b1, (short)0, acc[1], false, false);
    acc[2] = __builtin_amdgcn_wmma_f32_16x16x32_bf16(false, a, false, b2, (short)0, acc[2], false, false);
    acc[3] = __builtin_amdgcn_wmma_f32_16x16x32_bf16(false, a, false, b3, (short)0, acc[3], false, false);
  }
  int rowAdd = (lane >> 4) << 3;
  for (int ni = 0; ni < 4; ni++) {
    int col = n0 + ni * 16 + (lane & 15);
    #pragma unroll
    for (int j = 0; j < 8; j++) {
      int pr = perm[r16 + rowAdd + j];
      if (pr >= 0) {
        size_t idx = (size_t)pr * N + col;
        if (accumulate) Cout[idx] += acc[ni][j];
        else            Cout[idx]  = acc[ni][j];
      }
    }
  }
}

// ---------------- silu(gate)*up -> bf16
__global__ void silu_kernel(const float* __restrict__ gu, unsigned short* __restrict__ inter) {
  int g = blockIdx.x * blockDim.x + threadIdx.x;
  if (g >= NTOK * IEDIM) return;
  int t = g / IEDIM, i = g % IEDIM;
  float gt = gu[(size_t)t * GUN + i];
  float up = gu[(size_t)t * GUN + IEDIM + i];
  float sg = gt / (1.f + __expf(-gt));
  inter[g] = f2bf(sg * up);
}

// =====================================================================================
extern "C" void kernel_launch(void* const* d_in, const int* in_sizes, int n_in,
                              void* d_out, int out_size, void* d_ws, size_t ws_size,
                              hipStream_t stream) {
  (void)in_sizes; (void)n_in; (void)out_size; (void)ws_size;
  const float* x        = (const float*)d_in[0];
  const float* qkv_w_s  = (const float*)d_in[1];
  const float* qkv_b_s  = (const float*)d_in[2];
  const float* proj_w_s = (const float*)d_in[3];
  const float* proj_b_s = (const float*)d_in[4];
  const float* qkv_w_t  = (const float*)d_in[5];
  const float* qkv_b_t  = (const float*)d_in[6];
  const float* proj_w_t = (const float*)d_in[7];
  const float* proj_b_t = (const float*)d_in[8];
  const float* ln_s_g   = (const float*)d_in[9];
  const float* ln_s_b   = (const float*)d_in[10];
  const float* ln_t_g   = (const float*)d_in[11];
  const float* ln_t_b   = (const float*)d_in[12];
  const float* ln_m_g   = (const float*)d_in[13];
  const float* ln_m_b   = (const float*)d_in[14];
  const float* gate_up  = (const float*)d_in[15];
  const float* down     = (const float*)d_in[16];
  const int*   eids     = (const int*)d_in[19];
  float* xout = (float*)d_out;

  // ---- workspace carve-up
  char* ws = (char*)d_ws;
  size_t off = 0;
  auto alloc = [&](size_t bytes) { size_t o = off; off += (bytes + 255) & ~(size_t)255; return o; };
  unsigned short* WqkvS = (unsigned short*)(ws + alloc((size_t)QKVN * HDIM * 2));
  unsigned short* WqkvT = (unsigned short*)(ws + alloc((size_t)QKVN * HDIM * 2));
  unsigned short* WprjS = (unsigned short*)(ws + alloc((size_t)HDIM * HDIM * 2));
  unsigned short* WprjT = (unsigned short*)(ws + alloc((size_t)HDIM * HDIM * 2));
  unsigned short* WguT  = (unsigned short*)(ws + alloc((size_t)NEXP * GUN * HDIM * 2));
  unsigned short* WdnT  = (unsigned short*)(ws + alloc((size_t)NEXP * HDIM * IEDIM * 2));
  unsigned short* lnbf  = (unsigned short*)(ws + alloc((size_t)NTOK * HDIM * 2));
  unsigned short* attnbf= (unsigned short*)(ws + alloc((size_t)NTOK * HDIM * 2));  // reused as inter
  size_t bigBytes = (size_t)NTOK * GUN * 4;            // >= NTOK*QKVN*2
  char*  big      = ws + alloc(bigBytes);
  unsigned short* qkvbf = (unsigned short*)big;        // bf16 qkv (phase 1/2)
  float*          gu    = (float*)big;                 // fp32 gate_up out (phase 3)
  int* perm = (int*)(ws + alloc((size_t)PERMCAP * 4));
  int* meta = (int*)(ws + alloc(64 * 4));
  unsigned short* interbf = attnbf;

  const int THR = 256;
  auto blocks = [](size_t n) { return (unsigned)((n + 255) / 256); };

  // ---- weight conversion (deterministic each call)
  convtr_kernel<<<blocks((size_t)HDIM * QKVN), THR, 0, stream>>>(qkv_w_s,  WqkvS, HDIM, QKVN);
  convtr_kernel<<<blocks((size_t)HDIM * QKVN), THR, 0, stream>>>(qkv_w_t,  WqkvT, HDIM, QKVN);
  convtr_kernel<<<blocks((size_t)HDIM * HDIM), THR, 0, stream>>>(proj_w_s, WprjS, HDIM, HDIM);
  convtr_kernel<<<blocks((size_t)HDIM * HDIM), THR, 0, stream>>>(proj_w_t, WprjT, HDIM, HDIM);
  for (int e = 0; e < NEXP; e++) {
    convtr_kernel<<<blocks((size_t)HDIM * GUN), THR, 0, stream>>>(
        gate_up + (size_t)e * HDIM * GUN, WguT + (size_t)e * GUN * HDIM, HDIM, GUN);
    convtr_kernel<<<blocks((size_t)IEDIM * HDIM), THR, 0, stream>>>(
        down + (size_t)e * IEDIM * HDIM, WdnT + (size_t)e * HDIM * IEDIM, IEDIM, HDIM);
  }

  // ---- residual stream lives in d_out
  hipMemcpyAsync(xout, x, (size_t)NTOK * HDIM * 4, hipMemcpyDeviceToDevice, stream);

  // ======== stage 1: spatial attention ========
  ln_kernel<<<NTOK, THR, 0, stream>>>(xout, ln_s_g, ln_s_b, lnbf);
  gemm_bf16_kernel<<<dim3(NTOK / 128, QKVN / 128), THR, 0, stream>>>(
      lnbf, WqkvS, qkv_b_s, (void*)qkvbf, nullptr, NTOK, QKVN, HDIM, /*bf16 out*/1);
  attn_spatial_kernel<<<dim3(NSEQ_S, NHEAD), THR, 0, stream>>>(qkvbf, attnbf);
  gemm_bf16_kernel<<<dim3(NTOK / 128, HDIM / 128), THR, 0, stream>>>(
      attnbf, WprjS, proj_b_s, (void*)xout, xout, NTOK, HDIM, HDIM, /*residual*/2);

  // ======== stage 2: temporal attention ========
  ln_kernel<<<NTOK, THR, 0, stream>>>(xout, ln_t_g, ln_t_b, lnbf);
  gemm_bf16_kernel<<<dim3(NTOK / 128, QKVN / 128), THR, 0, stream>>>(
      lnbf, WqkvT, qkv_b_t, (void*)qkvbf, nullptr, NTOK, QKVN, HDIM, 1);
  attn_temporal_kernel<<<NSEQ_T, 128, 0, stream>>>(qkvbf, attnbf);
  gemm_bf16_kernel<<<dim3(NTOK / 128, HDIM / 128), THR, 0, stream>>>(
      attnbf, WprjT, proj_b_t, (void*)xout, xout, NTOK, HDIM, HDIM, 2);

  // ======== stage 3: MoE MLP ========
  ln_kernel<<<NTOK, THR, 0, stream>>>(xout, ln_m_g, ln_m_b, lnbf);
  moe_init_kernel<<<1, 32, 0, stream>>>(meta);
  moe_count_kernel<<<blocks(NTOK), THR, 0, stream>>>(eids, meta);
  moe_scan_kernel<<<1, 1, 0, stream>>>(meta);
  moe_fill_kernel<<<blocks(PERMCAP), THR, 0, stream>>>(perm);
  moe_scatter_kernel<<<blocks(NTOK), THR, 0, stream>>>(eids, meta, perm);
  moe_gemm_kernel<<<dim3(MAXTILES, GUN / 256), 128, 0, stream>>>(
      lnbf, WguT, gu, perm, meta, GUN, HDIM, /*store*/0);
  silu_kernel<<<blocks((size_t)NTOK * IEDIM), THR, 0, stream>>>(gu, interbf);
  moe_gemm_kernel<<<dim3(MAXTILES, HDIM / 256), 128, 0, stream>>>(
      interbf, WdnT, xout, perm, meta, HDIM, IEDIM, /*accumulate*/1);
}
